// LSTM_15204184228186
// MI455X (gfx1250) — matrix operand order repaired and checked by reading, not compiled, not run on hardware
//
#include <hip/hip_runtime.h>
#include <hip/hip_bf16.h>

typedef __bf16 bf16_t;
typedef __attribute__((ext_vector_type(16))) __bf16 v16bf;
typedef __attribute__((ext_vector_type(8)))  __bf16 v8bf;
typedef __attribute__((ext_vector_type(8)))  float  v8f;

namespace {
constexpr int kB     = 128;           // batch
constexpr int kS     = 512;           // encoder sequence length
constexpr int kD     = 64;            // model in/out dim
constexpr int kH     = 512;           // hidden dim
constexpr int kG     = 4 * kH;        // gate rows (i,f,g,o)
constexpr int kLabel = 96;
constexpr int kPred  = 192;
constexpr int kDecT  = kLabel + kPred; // 288 (x_dec time length)
}

__device__ __forceinline__ float fsigm(float x) { return 1.0f / (1.0f + __expf(-x)); }
__device__ __forceinline__ float ftanh(float x) {
  float e = __expf(-2.0f * x);
  return (1.0f - e) / (1.0f + e);
}

// A-fragment (16x32 bf16): lane m = lane&15 holds K chunks
// [halfSel*16, +8) and [halfSel*16+8, +8) where halfSel = lane>>4.
__device__ __forceinline__ v16bf load_a_frag(const bf16_t* row, int k0, int halfSel) {
  const v8bf lo = *(const v8bf*)(row + k0 + halfSel * 16);
  const v8bf hi = *(const v8bf*)(row + k0 + halfSel * 16 + 8);
  v16bf a;
#pragma unroll
  for (int i = 0; i < 8; ++i) { a[i] = lo[i]; a[i + 8] = hi[i]; }
  return a;
}

// ---------------------------------------------------------------------------
// One LSTM layer, one timestep:  G = X*Wih^T + Hprev*Whh^T + b;  c,h update.
// Grid: 16 blocks x 256 threads (8 waves) = 128 waves.
// Wave w: rp = w>>5 (0..3) -> batch rows [rp*32, rp*32+32) as TWO 16-row
// tiles, ct = w&31 -> hidden cols [ct*16, ct*16+16). Each weight (B) fragment
// feeds 2 WMMAs (both row tiles), halving L2 weight traffic on the serial
// critical path. 8 f32 accumulators = 4 gates x 2 row tiles.
// ---------------------------------------------------------------------------
__global__ __launch_bounds__(256)
void k_lstm_step(const bf16_t* __restrict__ X, int ldx, int Din,
                 const bf16_t* __restrict__ Hprev,
                 const float* Cprev,
                 const bf16_t* __restrict__ Wih,   // [4H, Din] bf16 row-major
                 const bf16_t* __restrict__ Whh,   // [4H, H]  bf16 row-major
                 const float* __restrict__ bias,   // [4H] = bih + bhh
                 bf16_t* __restrict__ Hout,        // [B, H] bf16
                 float* Cout)                      // [B, H] f32 (may alias Cprev)
{
  const int lane    = threadIdx.x & 31;
  const int wid     = threadIdx.x >> 5;
  const int w       = blockIdx.x * 8 + wid;   // 0..127
  const int rp      = w >> 5;                 // 0..3 (pair of row tiles)
  const int ct      = w & 31;                 // 0..31
  const int halfSel = lane >> 4;              // 0|1
  const int nIdx    = lane & 15;

  // Accumulators start at the (column-only) bias: C/D layout has n = lane&15.
  v8f acc[2][4];
#pragma unroll
  for (int q = 0; q < 4; ++q) {
    const float bq = bias[q * kH + ct * 16 + nIdx];
#pragma unroll
    for (int i = 0; i < 8; ++i) { acc[0][q][i] = bq; acc[1][q][i] = bq; }
  }

  const int am0 = rp * 32 + nIdx;       // A row, tile 0 (batch index)
  const int am1 = am0 + 16;             // A row, tile 1

  // ---- X @ Wih^T :  B[k,n] = Wih[n_global, k]  (row-major row = column op) ----
  {
    const bf16_t* xrow0 = X + (size_t)am0 * ldx;
    const bf16_t* xrow1 = X + (size_t)am1 * ldx;
    const bf16_t* wrow[4];
#pragma unroll
    for (int q = 0; q < 4; ++q)
      wrow[q] = Wih + (size_t)(q * kH + ct * 16 + nIdx) * Din;
    for (int k0 = 0; k0 < Din; k0 += 32) {
      const v16bf a0 = load_a_frag(xrow0, k0, halfSel);
      const v16bf a1 = load_a_frag(xrow1, k0, halfSel);
#pragma unroll
      for (int q = 0; q < 4; ++q) {
        const v16bf b = *(const v16bf*)(wrow[q] + k0 + halfSel * 16);
        acc[0][q] = __builtin_amdgcn_wmma_f32_16x16x32_bf16(
            false, a0, false, b, (short)0, acc[0][q], false, false);
        acc[1][q] = __builtin_amdgcn_wmma_f32_16x16x32_bf16(
            false, a1, false, b, (short)0, acc[1][q], false, false);
      }
    }
  }

  // ---- Hprev @ Whh^T ----
  {
    const bf16_t* hrow0 = Hprev + (size_t)am0 * kH;
    const bf16_t* hrow1 = Hprev + (size_t)am1 * kH;
    const bf16_t* wrow[4];
#pragma unroll
    for (int q = 0; q < 4; ++q)
      wrow[q] = Whh + (size_t)(q * kH + ct * 16 + nIdx) * kH;
    for (int k0 = 0; k0 < kH; k0 += 32) {
      const v16bf a0 = load_a_frag(hrow0, k0, halfSel);
      const v16bf a1 = load_a_frag(hrow1, k0, halfSel);
#pragma unroll
      for (int q = 0; q < 4; ++q) {
        const v16bf b = *(const v16bf*)(wrow[q] + k0 + halfSel * 16);
        acc[0][q] = __builtin_amdgcn_wmma_f32_16x16x32_bf16(
            false, a0, false, b, (short)0, acc[0][q], false, false);
        acc[1][q] = __builtin_amdgcn_wmma_f32_16x16x32_bf16(
            false, a1, false, b, (short)0, acc[1][q], false, false);
      }
    }
  }

  // ---- elementwise LSTM cell update on the two 16x16 tiles (C/D layout) ----
  const int col = ct * 16 + nIdx;
#pragma unroll
  for (int rt = 0; rt < 2; ++rt) {
#pragma unroll
    for (int v = 0; v < 8; ++v) {
      const int row    = rp * 32 + rt * 16 + v + halfSel * 8;
      const size_t idx = (size_t)row * kH + col;
      const float cp = Cprev[idx];
      const float ig = fsigm(acc[rt][0][v]);
      const float fg = fsigm(acc[rt][1][v]);
      const float gg = ftanh(acc[rt][2][v]);
      const float og = fsigm(acc[rt][3][v]);
      const float c  = fg * cp + ig * gg;
      const float h  = og * ftanh(c);
      Cout[idx] = c;
      Hout[idx] = (bf16_t)h;
    }
  }
}

// ---------------------------------------------------------------------------
// Projection: out[b,d] = h1[b,:] . projW[d,:] + proj_b[d]   (128x64, K=512)
// Writes f32 slice of d_out (row stride kPred*kD) and bf16 next-step input.
// Grid: 4 blocks x 256 threads = 32 waves = 8 row tiles x 4 col tiles.
// ---------------------------------------------------------------------------
__global__ __launch_bounds__(256)
void k_proj(const bf16_t* __restrict__ Hin,   // [B, H] bf16
            const bf16_t* __restrict__ Wp,    // [D, H] bf16
            const float* __restrict__ bp,     // [D] f32
            float* __restrict__ outp,         // d_out + s*kD
            bf16_t* __restrict__ pred)        // [B, D] bf16
{
  const int lane    = threadIdx.x & 31;
  const int wid     = threadIdx.x >> 5;
  const int w       = blockIdx.x * 8 + wid;   // 0..31
  const int rt      = w >> 2;                 // 0..7
  const int ct      = w & 3;                  // 0..3
  const int halfSel = lane >> 4;
  const int nIdx    = lane & 15;

  v8f acc;
  {
    const float bq = bp[ct * 16 + nIdx];
#pragma unroll
    for (int i = 0; i < 8; ++i) acc[i] = bq;
  }

  const bf16_t* hrow = Hin + (size_t)(rt * 16 + nIdx) * kH;
  const bf16_t* wrow = Wp + (size_t)(ct * 16 + nIdx) * kH;
  for (int k0 = 0; k0 < kH; k0 += 32) {
    const v16bf a = load_a_frag(hrow, k0, halfSel);
    const v16bf b = *(const v16bf*)(wrow + k0 + halfSel * 16);
    acc = __builtin_amdgcn_wmma_f32_16x16x32_bf16(
        false, a, false, b, (short)0, acc, false, false);
  }

  const int col = ct * 16 + nIdx;
#pragma unroll
  for (int v = 0; v < 8; ++v) {
    const int row = rt * 16 + v + halfSel * 8;
    outp[(size_t)row * (kPred * kD) + col] = acc[v];
    pred[row * kD + col] = (bf16_t)acc[v];
  }
}

// ---------------------------------------------------------------------------
// One-time prep kernels
// ---------------------------------------------------------------------------
__global__ void k_f2bf(const float* __restrict__ src, bf16_t* __restrict__ dst, int n) {
  const int i = blockIdx.x * blockDim.x + threadIdx.x;
  if (i < n) dst[i] = (bf16_t)src[i];
}

__global__ void k_addbias(const float* __restrict__ a, const float* __restrict__ b,
                          float* __restrict__ d, int n) {
  const int i = blockIdx.x * blockDim.x + threadIdx.x;
  if (i < n) d[i] = a[i] + b[i];
}

// ---------------------------------------------------------------------------
extern "C" void kernel_launch(void* const* d_in, const int* in_sizes, int n_in,
                              void* d_out, int out_size, void* d_ws, size_t ws_size,
                              hipStream_t stream) {
  (void)in_sizes; (void)n_in; (void)out_size; (void)ws_size;

  // ---- workspace carve-out ----
  char* p = (char*)d_ws;
  auto alloc = [&](size_t bytes) -> void* {
    void* r = (void*)p;
    p += (bytes + 255) & ~(size_t)255;
    return r;
  };

  bf16_t* eWih0 = (bf16_t*)alloc((size_t)kG * kD * 2);
  bf16_t* eWhh0 = (bf16_t*)alloc((size_t)kG * kH * 2);
  bf16_t* eWih1 = (bf16_t*)alloc((size_t)kG * kH * 2);
  bf16_t* eWhh1 = (bf16_t*)alloc((size_t)kG * kH * 2);
  bf16_t* dWih0 = (bf16_t*)alloc((size_t)kG * kD * 2);
  bf16_t* dWhh0 = (bf16_t*)alloc((size_t)kG * kH * 2);
  bf16_t* dWih1 = (bf16_t*)alloc((size_t)kG * kH * 2);
  bf16_t* dWhh1 = (bf16_t*)alloc((size_t)kG * kH * 2);
  bf16_t* pW    = (bf16_t*)alloc((size_t)kD * kH * 2);
  float*  eB0   = (float*)alloc((size_t)kG * 4);
  float*  eB1   = (float*)alloc((size_t)kG * 4);
  float*  dB0   = (float*)alloc((size_t)kG * 4);
  float*  dB1   = (float*)alloc((size_t)kG * 4);
  bf16_t* xe    = (bf16_t*)alloc((size_t)kB * kS * kD * 2);
  bf16_t* xd    = (bf16_t*)alloc((size_t)kB * kDecT * kD * 2);
  bf16_t* h0a   = (bf16_t*)alloc((size_t)kB * kH * 2);
  bf16_t* h0b   = (bf16_t*)alloc((size_t)kB * kH * 2);
  bf16_t* h1a   = (bf16_t*)alloc((size_t)kB * kH * 2);
  bf16_t* h1b   = (bf16_t*)alloc((size_t)kB * kH * 2);
  float*  c0    = (float*)alloc((size_t)kB * kH * 4);
  float*  c1    = (float*)alloc((size_t)kB * kH * 4);
  bf16_t* predb = (bf16_t*)alloc((size_t)kB * kD * 2);

  // ---- one-time conversions (weights/inputs -> bf16, biases combined) ----
  auto conv = [&](const void* s, bf16_t* dst, size_t n) {
    k_f2bf<<<(unsigned)((n + 255) / 256), 256, 0, stream>>>((const float*)s, dst, (int)n);
  };
  conv(d_in[4],  eWih0, (size_t)kG * kD);
  conv(d_in[5],  eWhh0, (size_t)kG * kH);
  conv(d_in[8],  eWih1, (size_t)kG * kH);
  conv(d_in[9],  eWhh1, (size_t)kG * kH);
  conv(d_in[12], dWih0, (size_t)kG * kD);
  conv(d_in[13], dWhh0, (size_t)kG * kH);
  conv(d_in[16], dWih1, (size_t)kG * kH);
  conv(d_in[17], dWhh1, (size_t)kG * kH);
  conv(d_in[20], pW,    (size_t)kD * kH);
  conv(d_in[0],  xe,    (size_t)kB * kS * kD);
  conv(d_in[2],  xd,    (size_t)kB * kDecT * kD);

  auto bsum = [&](const void* a, const void* b, float* dst) {
    k_addbias<<<(kG + 255) / 256, 256, 0, stream>>>((const float*)a, (const float*)b, dst, kG);
  };
  bsum(d_in[6],  d_in[7],  eB0);
  bsum(d_in[10], d_in[11], eB1);
  bsum(d_in[14], d_in[15], dB0);
  bsum(d_in[18], d_in[19], dB1);

  // ---- zero-init states ----
  hipMemsetAsync(h0a, 0, (size_t)kB * kH * 2, stream);
  hipMemsetAsync(h1a, 0, (size_t)kB * kH * 2, stream);
  hipMemsetAsync(c0,  0, (size_t)kB * kH * 4, stream);
  hipMemsetAsync(c1,  0, (size_t)kB * kH * 4, stream);

  bf16_t* h0c = h0a; bf16_t* h0n = h0b;
  bf16_t* h1c = h1a; bf16_t* h1n = h1b;

  auto step0 = [&](const bf16_t* X, int ldx, int Din,
                   const bf16_t* Wih, const bf16_t* Whh, const float* bias) {
    k_lstm_step<<<16, 256, 0, stream>>>(X, ldx, Din, h0c, c0, Wih, Whh, bias, h0n, c0);
    bf16_t* t = h0c; h0c = h0n; h0n = t;
  };
  auto step1 = [&](const bf16_t* Wih, const bf16_t* Whh, const float* bias) {
    k_lstm_step<<<16, 256, 0, stream>>>(h0c, kH, kH, h1c, c1, Wih, Whh, bias, h1n, c1);
    bf16_t* t = h1c; h1c = h1n; h1n = t;
  };

  // ---- encoder: 512 steps ----
  for (int t = 0; t < kS; ++t) {
    step0(xe + (size_t)t * kD, kS * kD, kD, eWih0, eWhh0, eB0);
    step1(eWih1, eWhh1, eB1);
  }

  // ---- teacher-forced decode: 96 steps ----
  for (int t = 0; t < kLabel; ++t) {
    step0(xd + (size_t)t * kD, kDecT * kD, kD, dWih0, dWhh0, dB0);
    step1(dWih1, dWhh1, dB1);
  }

  // ---- pred0 ----
  const float* proj_b = (const float*)d_in[21];
  k_proj<<<4, 256, 0, stream>>>(h1c, pW, proj_b, (float*)d_out, predb);

  // ---- autoregressive decode: 191 steps ----
  for (int s = 1; s < kPred; ++s) {
    step0(predb, kD, kD, dWih0, dWhh0, dB0);
    step1(dWih1, dWhh1, dB1);
    k_proj<<<4, 256, 0, stream>>>(h1c, pW, proj_b, (float*)d_out + (size_t)s * kD, predb);
  }
}
